// MultiHeadAttentionLayer_59322088292536
// MI455X (gfx1250) — matrix operand order repaired and checked by reading, not compiled, not run on hardware
//
#include <hip/hip_runtime.h>

// ---------------------------------------------------------------------------
// MultiHeadAttentionLayer for MI455X (gfx1250): bf16 WMMA everywhere.
//   B=2, S=2048, H=1024, NH=16, HD=64
//   Wq/Wk/Wv -> bf16 once; Q = query @ Wq^T + bq (likewise K,V) -> bf16 in ws
//   per-head (contiguous-slice reshape) flash attention with online softmax
// ---------------------------------------------------------------------------

typedef __bf16 bf16_t;
typedef __attribute__((ext_vector_type(16))) __bf16 v16bf;
typedef __attribute__((ext_vector_type(8)))  __bf16 v8bf;
typedef __attribute__((ext_vector_type(8)))  float  v8f;
typedef __attribute__((ext_vector_type(4)))  float  v4f;

#define DIM_B  2
#define DIM_S  2048
#define DIM_H  1024
#define DIM_NH 16
#define DIM_HD 64
#define M_TOT  (DIM_B * DIM_S)      // 4096 rows for the projections
#define CK     32                   // keys per flash-attention chunk
#define SQ     128                  // q rows per block (8 waves * 16)

// ---------------------------------------------------------------------------
// fp32 -> bf16 bulk convert (for weights). n must be a multiple of 2048.
// ---------------------------------------------------------------------------
__global__ __launch_bounds__(256)
void cvt_kernel(const float* __restrict__ x, bf16_t* __restrict__ y)
{
    const size_t i = ((size_t)blockIdx.x * 256 + threadIdx.x) * 8;
    const v4f a = *(const v4f*)(x + i);
    const v4f b = *(const v4f*)(x + i + 4);
    v8bf o;
#pragma unroll
    for (int j = 0; j < 4; ++j) { o[j] = (__bf16)a[j]; o[4 + j] = (__bf16)b[j]; }
    *(v8bf*)(y + i) = o;
}

// ---------------------------------------------------------------------------
// Projection: Y[m,n] = sum_k X[m,k] * Wb[n,k] + bias[n], output bf16.
// X fp32 (converted in-register), Wb bf16. Grid: (H/64, M/256), 8 waves/block,
// each wave computes a 32(M) x 64(N) tile = 8 accumulators, 8 WMMA per k-step.
// ---------------------------------------------------------------------------
__global__ __launch_bounds__(256)
void proj_kernel(const float* __restrict__ X, const bf16_t* __restrict__ Wb,
                 const float* __restrict__ bias, bf16_t* __restrict__ Y)
{
    const int lane = threadIdx.x & 31;
    const int wave = threadIdx.x >> 5;
    const int h    = lane >> 4;       // half (0/1)
    const int ln   = lane & 15;
    const int n0   = blockIdx.x * 64;
    const int m0   = blockIdx.y * 256 + wave * 32;

    // C/D layout: element r of v8f = (m = r + 8*h, n = ln). Bias depends on n.
    v8f acc[4][2];
#pragma unroll
    for (int t = 0; t < 4; ++t) {
        const float bv = bias[n0 + t * 16 + ln];
#pragma unroll
        for (int u = 0; u < 2; ++u)
#pragma unroll
            for (int r = 0; r < 8; ++r) acc[t][u][r] = bv;
    }

    const float* arow0 = X + (size_t)(m0 + ln) * DIM_H;
    const float* arow1 = X + (size_t)(m0 + 16 + ln) * DIM_H;

    for (int k0 = 0; k0 < DIM_H; k0 += 32) {
        // A fragments 16x32: lane = row, k-runs [k0+8h,+8) and [k0+16+8h,+8)
        v16bf a[2];
#pragma unroll
        for (int u = 0; u < 2; ++u) {
            const float* p0 = (u ? arow1 : arow0) + k0 + 8 * h;
            const float* p1 = (u ? arow1 : arow0) + k0 + 16 + 8 * h;
            const v4f x0 = *(const v4f*)p0, x1 = *(const v4f*)(p0 + 4);
            const v4f x2 = *(const v4f*)p1, x3 = *(const v4f*)(p1 + 4);
#pragma unroll
            for (int i = 0; i < 4; ++i) {
                a[u][i]      = (__bf16)x0[i];
                a[u][4 + i]  = (__bf16)x1[i];
                a[u][8 + i]  = (__bf16)x2[i];
                a[u][12 + i] = (__bf16)x3[i];
            }
        }
#pragma unroll
        for (int t = 0; t < 4; ++t) {
            // B fragment 32x16 (bf16 direct): lane = column n0+16t+ln,
            // k-runs [k0+8h,+8) and [k0+16+8h,+8) from weight row.
            const bf16_t* wrow = Wb + (size_t)(n0 + t * 16 + ln) * DIM_H + k0 + 8 * h;
            const v8bf w0 = *(const v8bf*)wrow;
            const v8bf w1 = *(const v8bf*)(wrow + 16);
            v16bf b;
#pragma unroll
            for (int i = 0; i < 8; ++i) { b[i] = w0[i]; b[8 + i] = w1[i]; }
            acc[t][0] = __builtin_amdgcn_wmma_f32_16x16x32_bf16(
                false, a[0], false, b, (short)0, acc[t][0], false, false);
            acc[t][1] = __builtin_amdgcn_wmma_f32_16x16x32_bf16(
                false, a[1], false, b, (short)0, acc[t][1], false, false);
        }
    }

#pragma unroll
    for (int t = 0; t < 4; ++t) {
#pragma unroll
        for (int u = 0; u < 2; ++u) {
#pragma unroll
            for (int r = 0; r < 8; ++r) {
                const int mm = m0 + u * 16 + r + 8 * h;
                const int nn = n0 + t * 16 + ln;
                Y[(size_t)mm * DIM_H + nn] = (bf16_t)acc[t][u][r];
            }
        }
    }
}

// ---------------------------------------------------------------------------
// Flash attention per (batch, head): Qh,Kh,Vh are [2048,64] bf16 contiguous
// slices. Grid: (S/SQ, NH, B), block 256 = 8 waves, 16 q-rows per wave.
// ---------------------------------------------------------------------------
__global__ __launch_bounds__(256)
void attn_kernel(const bf16_t* __restrict__ Qb, const bf16_t* __restrict__ Kb,
                 const bf16_t* __restrict__ Vb, float* __restrict__ out)
{
    const int lane  = threadIdx.x & 31;
    const int wave  = threadIdx.x >> 5;
    const int h     = lane >> 4;
    const int ln    = lane & 15;
    const int head  = blockIdx.y;
    const int batch = blockIdx.z;

    const size_t hoff = (size_t)batch * DIM_S * DIM_H + (size_t)head * (DIM_S * DIM_HD);
    const bf16_t* Qh = Qb + hoff;
    const bf16_t* Kh = Kb + hoff;
    const bf16_t* Vh = Vb + hoff;

    const int q0 = blockIdx.x * SQ + wave * 16;

    // LDS staging (row strides are multiples of 8 elems = 16B for alignment)
    __shared__ __align__(16) bf16_t Kch[CK][DIM_HD + 8];      // 32 x 72
    __shared__ __align__(16) bf16_t Vt[DIM_HD][CK + 8];       // 64 x 40 (transposed)
    __shared__ __align__(16) bf16_t Pl[8][16][CK + 8];        // per-wave P tile

    // Q fragments (d = 0..31, 32..63), loaded once.
    v16bf qa[2];
    {
        const bf16_t* qrow = Qh + (size_t)(q0 + ln) * DIM_HD;
#pragma unroll
        for (int f = 0; f < 2; ++f) {
            const v8bf lo = *(const v8bf*)(qrow + f * 32 + 16 * h);
            const v8bf hi = *(const v8bf*)(qrow + f * 32 + 16 * h + 8);
#pragma unroll
            for (int i = 0; i < 8; ++i) { qa[f][i] = lo[i]; qa[f][8 + i] = hi[i]; }
        }
    }

    v8f o[4] = {v8f{}, v8f{}, v8f{}, v8f{}};
    float mrow[8], lrow[8];
#pragma unroll
    for (int r = 0; r < 8; ++r) { mrow[r] = -1.0e30f; lrow[r] = 0.0f; }

    for (int kk0 = 0; kk0 < DIM_S; kk0 += CK) {
        // ---- cooperative staging: K row-major, V transposed -----------------
        {
            const int row  = threadIdx.x >> 3;        // 0..31 key row
            const int col8 = (threadIdx.x & 7) * 8;   // 0..56 head-dim col
            *(v8bf*)&Kch[row][col8] =
                *(const v8bf*)(Kh + (size_t)(kk0 + row) * DIM_HD + col8);
            const v8bf vv = *(const v8bf*)(Vh + (size_t)(kk0 + row) * DIM_HD + col8);
#pragma unroll
            for (int i = 0; i < 8; ++i) Vt[col8 + i][row] = vv[i];
        }
        if (kk0 + CK < DIM_S) {
            __builtin_prefetch(Kh + (size_t)(kk0 + CK) * DIM_HD, 0, 0);
            __builtin_prefetch(Vh + (size_t)(kk0 + CK) * DIM_HD, 0, 0);
        }
        __syncthreads();

        // ---- S = (Q K^T) * 1/sqrt(64) for two 16-key subtiles ---------------
        v8f s[2];
#pragma unroll
        for (int j = 0; j < 2; ++j) {
            const bf16_t* kr = &Kch[j * 16 + ln][0];
            v16bf b0, b1;
#pragma unroll
            for (int i = 0; i < 16; ++i) b0[i] = kr[16 * h + i];        // d in [0,32)
#pragma unroll
            for (int i = 0; i < 16; ++i) b1[i] = kr[32 + 16 * h + i];   // d in [32,64)
            v8f z = v8f{};
            z    = __builtin_amdgcn_wmma_f32_16x16x32_bf16(false, qa[0], false, b0,
                                                           (short)0, z, false, false);
            s[j] = __builtin_amdgcn_wmma_f32_16x16x32_bf16(false, qa[1], false, b1,
                                                           (short)0, z, false, false);
        }
        const float scale = 0.125f;   // 1/sqrt(HD)
#pragma unroll
        for (int r = 0; r < 8; ++r) { s[0][r] *= scale; s[1][r] *= scale; }

        // ---- online softmax update -----------------------------------------
#pragma unroll
        for (int r = 0; r < 8; ++r) {
            float v = fmaxf(s[0][r], s[1][r]);
#pragma unroll
            for (int off = 1; off < 16; off <<= 1)
                v = fmaxf(v, __shfl_xor(v, off, 32));       // reduce within half
            const float mnew  = fmaxf(mrow[r], v);
            const float alpha = __expf(mrow[r] - mnew);
            mrow[r] = mnew;
            const float e0 = __expf(s[0][r] - mnew);
            const float e1 = __expf(s[1][r] - mnew);
            float rs = e0 + e1;
#pragma unroll
            for (int off = 1; off < 16; off <<= 1)
                rs += __shfl_xor(rs, off, 32);
            lrow[r] = lrow[r] * alpha + rs;
#pragma unroll
            for (int t = 0; t < 4; ++t) o[t][r] *= alpha;
            // P in C-layout element (m = r+8h, key = j*16+ln) -> LDS (m, key)
            Pl[wave][r + 8 * h][ln]      = (__bf16)e0;
            Pl[wave][r + 8 * h][16 + ln] = (__bf16)e1;
        }

        // ---- O += P * V ------------------------------------------------------
        v16bf pa;
        {
            const bf16_t* pr = &Pl[wave][ln][0];
#pragma unroll
            for (int i = 0; i < 8; ++i) pa[i]     = pr[16 * h + i];
#pragma unroll
            for (int i = 0; i < 8; ++i) pa[8 + i] = pr[16 * h + 8 + i];
        }
#pragma unroll
        for (int t = 0; t < 4; ++t) {
            const bf16_t* vr = &Vt[t * 16 + ln][0];   // column d = 16t+ln, keys contiguous
            v16bf vb;
#pragma unroll
            for (int i = 0; i < 16; ++i) vb[i] = vr[16 * h + i];
            o[t] = __builtin_amdgcn_wmma_f32_16x16x32_bf16(false, pa, false, vb,
                                                           (short)0, o[t], false, false);
        }
        __syncthreads();
    }

    // ---- normalize + scatter: out[b, q, head*64 + d] ------------------------
#pragma unroll
    for (int t = 0; t < 4; ++t) {
#pragma unroll
        for (int r = 0; r < 8; ++r) {
            const int q = q0 + r + 8 * h;
            const int d = t * 16 + ln;
            out[(size_t)batch * DIM_S * DIM_H + (size_t)q * DIM_H + head * DIM_HD + d] =
                o[t][r] / lrow[r];
        }
    }
}

// ---------------------------------------------------------------------------
extern "C" void kernel_launch(void* const* d_in, const int* in_sizes, int n_in,
                              void* d_out, int out_size, void* d_ws, size_t ws_size,
                              hipStream_t stream)
{
    const float* query = (const float*)d_in[0];
    const float* key   = (const float*)d_in[1];
    const float* value = (const float*)d_in[2];
    const float* Wq    = (const float*)d_in[3];
    const float* bq    = (const float*)d_in[4];
    const float* Wk    = (const float*)d_in[5];
    const float* bk    = (const float*)d_in[6];
    const float* Wv    = (const float*)d_in[7];
    const float* bv    = (const float*)d_in[8];
    float* out = (float*)d_out;

    // workspace layout (bf16): Q,K,V projections (3 x 4096*1024) + weights (3 x 1024*1024)
    bf16_t* Qbf = (bf16_t*)d_ws;
    bf16_t* Kbf = Qbf + (size_t)M_TOT * DIM_H;
    bf16_t* Vbf = Kbf + (size_t)M_TOT * DIM_H;
    bf16_t* Wqb = Vbf + (size_t)M_TOT * DIM_H;
    bf16_t* Wkb = Wqb + (size_t)DIM_H * DIM_H;
    bf16_t* Wvb = Wkb + (size_t)DIM_H * DIM_H;

    // 1) weights fp32 -> bf16 (one-time; kills redundant per-block conversion)
    const int cgrid = (DIM_H * DIM_H) / (256 * 8);   // 512 blocks
    cvt_kernel<<<cgrid, 256, 0, stream>>>(Wq, Wqb);
    cvt_kernel<<<cgrid, 256, 0, stream>>>(Wk, Wkb);
    cvt_kernel<<<cgrid, 256, 0, stream>>>(Wv, Wvb);

    // 2) projections (bf16 WMMA, f32 accumulate, bias seeded in C)
    const dim3 pgrid(DIM_H / 64, M_TOT / 256);
    proj_kernel<<<pgrid, 256, 0, stream>>>(query, Wqb, bq, Qbf);
    proj_kernel<<<pgrid, 256, 0, stream>>>(key,   Wkb, bk, Kbf);
    proj_kernel<<<pgrid, 256, 0, stream>>>(value, Wvb, bv, Vbf);

    // 3) flash attention per head
    const dim3 agrid(DIM_S / SQ, DIM_NH, DIM_B);
    attn_kernel<<<agrid, 256, 0, stream>>>(Qbf, Kbf, Vbf, out);
}
